// ScalarDotProductCriticNetworkRevised_18614388261539
// MI455X (gfx1250) — matrix operand order repaired and checked by reading, not compiled, not run on hardware
//
#include <hip/hip_runtime.h>
#include <hip/hip_bf16.h>

typedef __attribute__((ext_vector_type(16))) _Float16 v16h;
typedef __attribute__((ext_vector_type(8)))  float    v8f;

#define NAG   32
#define DOBS  128
#define DACT  16
#define OAIP  160   // 144 padded to 160 (5 k-steps of 32)
#define HID   256

// d_ws offsets in u32 units (packed f16 pair layout: dst[p*N+n] = {W[2p,n],W[2p+1,n]})
#define PK_KP  0        // W_key_proc   128x128 -> 8192 u32
#define PK_QP  8192     // W_query_proc
#define PK_VP  16384    // W_value_proc
#define PK_K2  24576    // W_key
#define PK_Q2  32768    // W_query
#define PK_EMB 40960    // W_embed 144(->160)x128 -> 10240 u32
#define PK_AV  51200    // W_attn_val
#define PK_F1  59392    // W_f1 128x256 -> 16384 u32

union FragU { v16h v; unsigned int u[8]; _Float16 h[16]; };

__device__ inline v8f wmmaf(v16h a, v16h b, v8f c) {
    return __builtin_amdgcn_wmma_f32_16x16x32_f16(false, a, false, b, (short)0, c,
                                                  false, false);
}

// A-fragment (16x32 f16) from row-major f16 LDS, row stride in halves (even).
// A layout: lanes 0-15 row m, pairs 0..7 (K=0..15); lanes 16-31 pairs 8..15 (K=16..31).
__device__ inline v16h ldA(const _Float16* base, int stride, int m0, int k0, int lane) {
    FragU f;
    const int r  = m0 + (lane & 15);
    const int hi = lane >> 4;
    const unsigned int* row = (const unsigned int*)(base + r * stride);
    const int pb = (k0 >> 1) + (hi << 3);
#pragma unroll
    for (int v = 0; v < 8; ++v) {
        const int q = v;
        f.u[v] = row[pb + q];
    }
    return f.v;
}

// B-fragment (32x16 f16) from pre-packed global weights [Kpairs][N] of u32.
// B layout used here (contiguous k, as ldA): lanes 0-15 hold K=0..15 (pairs 0..7), lanes 16-31 K=16..31 (pairs 8..15).
__device__ inline v16h ldB_pk(const unsigned int* __restrict__ wp, int N, int n0, int k0,
                              int lane) {
    FragU f;
    const int n  = n0 + (lane & 15);
    const int pb = (k0 >> 1) + ((lane >> 4) << 3);
#pragma unroll
    for (int v = 0; v < 8; ++v) f.u[v] = wp[(pb + v) * N + n];
    return f.v;
}

// B-fragment for B = X^T where X is row-major [rows][stride] f16 in LDS (contiguous pairs).
__device__ inline v16h ldB_tr(const _Float16* X, int stride, int n0, int k0, int lane) {
    FragU f;
    const int n  = n0 + (lane & 15);
    const unsigned int* row = (const unsigned int*)(X + n * stride);
    const int pb = (k0 >> 1) + ((lane >> 4) << 3);
#pragma unroll
    for (int v = 0; v < 8; ++v) f.u[v] = row[pb + v];
    return f.v;
}

// B-fragment from row-major K x N f16 LDS (gather 2 halves per VGPR).
__device__ inline v16h ldB_rm(const _Float16* X, int stride, int n0, int k0, int lane) {
    FragU f;
    const int n  = n0 + (lane & 15);
    const int kb = k0 + ((lane >> 4) << 4);
#pragma unroll
    for (int v = 0; v < 8; ++v) {
        const int k = kb + 2 * v;
        f.h[2 * v]     = X[k * stride + n];
        f.h[2 * v + 1] = X[(k + 1) * stride + n];
    }
    return f.v;
}

// D store (16x16 f32 acc): VGPR v -> row m0 + v + 8*(lane>=16), col n0 + lane%16
__device__ inline void stD_h(_Float16* dst, int stride, int m0, int n0, int lane, v8f acc) {
    const int rb = m0 + ((lane >> 4) << 3);
    const int c  = n0 + (lane & 15);
#pragma unroll
    for (int v = 0; v < 8; ++v) dst[(rb + v) * stride + c] = (_Float16)acc[v];
}
__device__ inline void stD_f(float* dst, int stride, int m0, int n0, int lane, v8f acc,
                             float scale) {
    const int rb = m0 + ((lane >> 4) << 3);
    const int c  = n0 + (lane & 15);
#pragma unroll
    for (int v = 0; v < 8; ++v) dst[(rb + v) * stride + c] = acc[v] * scale;
}

// C[32][128] = A[32][K] @ Bpacked[K][128]; wave w owns n-tile w, both m-tiles.
__device__ inline void gemm128(const _Float16* A, int strideA, const unsigned int* Bp,
                               _Float16* C, int ksteps, int w, int lane) {
    const int n0 = w << 4;
    v8f acc0 = {}, acc1 = {};
    for (int ks = 0; ks < ksteps; ++ks) {
        const int k0 = ks << 5;
        v16h b  = ldB_pk(Bp, 128, n0, k0, lane);
        v16h a0 = ldA(A, strideA, 0, k0, lane);
        v16h a1 = ldA(A, strideA, 16, k0, lane);
        acc0 = wmmaf(a0, b, acc0);
        acc1 = wmmaf(a1, b, acc1);
    }
    stD_h(C, 128, 0, n0, lane, acc0);
    stD_h(C, 128, 16, n0, lane, acc1);
}

// S[32][32] = (Q[32][128] @ K^T) / sqrt(128); waves 0..3 each own one 16x16 tile.
__device__ inline void scores32(const _Float16* Q, const _Float16* Kt, float* S, int w,
                                int lane) {
    if (w < 4) {
        const int m0 = (w >> 1) << 4, n0 = (w & 1) << 4;
        v8f acc = {};
#pragma unroll
        for (int ks = 0; ks < 4; ++ks) {
            const int k0 = ks << 5;
            v16h a = ldA(Q, 128, m0, k0, lane);
            v16h b = ldB_tr(Kt, 128, n0, k0, lane);
            acc = wmmaf(a, b, acc);
        }
        stD_f(S, 32, m0, n0, lane, acc, 0.08838834764831845f);  // 1/sqrt(128)
    }
}

__device__ inline void softmax32(float* S, _Float16* H, int tid) {
    if (tid < 32) {
        float* row = S + tid * 32;
        float mx = row[0];
        for (int j = 1; j < 32; ++j) mx = fmaxf(mx, row[j]);
        float sum = 0.f;
        for (int j = 0; j < 32; ++j) {
            float e = __expf(row[j] - mx);
            row[j] = e;
            sum += e;
        }
        const float inv = 1.0f / sum;
        for (int j = 0; j < 32; ++j) {
            float v = row[j] * inv;
            row[j] = v;
            H[tid * 32 + j] = (_Float16)v;
        }
    }
}

__global__ __launch_bounds__(256)
void critic_fused_kernel(const float* __restrict__ states, const float* __restrict__ policies,
                         const float* __restrict__ actions, const float* __restrict__ wf2_g,
                         const unsigned int* __restrict__ ws, float* __restrict__ out_val,
                         float* __restrict__ out_w1, float* __restrict__ out_w2) {
    const int bid  = blockIdx.x;
    const int tid  = threadIdx.x;
    const int w    = tid >> 5;
    const int lane = tid & 31;

    __shared__ alignas(16) _Float16 sOA[NAG][OAIP];   // [states|actions|0pad]
    __shared__ alignas(16) _Float16 sOP[NAG][OAIP];   // [states|policies|0pad]
    __shared__ alignas(16) _Float16 sQ1[NAG][128];    // later: avA
    __shared__ alignas(16) _Float16 sK1[NAG][128];    // later: delta
    __shared__ alignas(16) _Float16 sV1[NAG][128];    // later: embA
    __shared__ alignas(16) _Float16 sQ2[NAG][128];    // later: embP
    __shared__ alignas(16) _Float16 sK2[NAG][128];
    __shared__ alignas(16) _Float16 sAgg[NAG][128];   // later: base
    __shared__ alignas(16) float    sS1[NAG][32];
    __shared__ alignas(16) float    sS2[NAG][32];
    __shared__ alignas(16) _Float16 sW1h[NAG][32];
    __shared__ alignas(16) _Float16 sW2h[NAG][32];
    __shared__ alignas(16) float    sPf[NAG][258];    // pad 258: conflict-free float2
    __shared__ alignas(16) float    sQf[NAG][258];
    __shared__ alignas(16) float    sF2[HID];
#define sAvA  sQ1
#define sDlt  sK1
#define sEmbA sV1
#define sEmbP sQ2
#define sBase sAgg

    // ---- Phase 0: load + convert inputs ----
    {
        const float* st = states + (size_t)bid * NAG * DOBS;
        const float* ac = actions + (size_t)bid * NAG * DACT;
        const float* po = policies + (size_t)bid * NAG * DACT;
        for (int i = tid; i < NAG * DOBS; i += 256) {
            const int r = i >> 7, c = i & 127;
            _Float16 h = (_Float16)st[i];
            sOA[r][c] = h;
            sOP[r][c] = h;
        }
        for (int i = tid; i < NAG * DACT; i += 256) {
            const int r = i >> 4, c = i & 15;
            sOA[r][DOBS + c] = (_Float16)ac[i];
            sOP[r][DOBS + c] = (_Float16)po[i];
        }
        for (int i = tid; i < NAG * 16; i += 256) {   // zero K pad 144..159
            const int r = i >> 4, c = i & 15;
            sOA[r][144 + c] = (_Float16)0.f;
            sOP[r][144 + c] = (_Float16)0.f;
        }
        sF2[tid] = wf2_g[tid];   // 256 threads, HID==256
    }
    __syncthreads();

    // ---- Phase 1: q1,k1,v1 = states @ W  (A rows live in sOA[:, :128]) ----
    gemm128(&sOA[0][0], OAIP, ws + PK_QP, &sQ1[0][0], 4, w, lane);
    gemm128(&sOA[0][0], OAIP, ws + PK_KP, &sK1[0][0], 4, w, lane);
    gemm128(&sOA[0][0], OAIP, ws + PK_VP, &sV1[0][0], 4, w, lane);
    __syncthreads();

    // ---- Phase 2: s1 = q1 @ k1^T / sqrt(128) ----
    scores32(&sQ1[0][0], &sK1[0][0], &sS1[0][0], w, lane);
    __syncthreads();
    softmax32(&sS1[0][0], &sW1h[0][0], tid);
    __syncthreads();
    for (int i = tid; i < NAG * NAG; i += 256)
        out_w1[(size_t)bid * NAG * NAG + i] = ((const float*)sS1)[i];

    // ---- Phase 3: agg = w1 @ v1  (K=32, one k-step) ----
    {
        const int n0 = w << 4;
        v16h b  = ldB_rm(&sV1[0][0], 128, n0, 0, lane);
        v16h a0 = ldA(&sW1h[0][0], 32, 0, 0, lane);
        v16h a1 = ldA(&sW1h[0][0], 32, 16, 0, lane);
        v8f c0 = {}, c1 = {};
        c0 = wmmaf(a0, b, c0);
        c1 = wmmaf(a1, b, c1);
        stD_h(&sAgg[0][0], 128, 0, n0, lane, c0);
        stD_h(&sAgg[0][0], 128, 16, n0, lane, c1);
    }
    __syncthreads();

    // ---- Phase 4: q2 = agg @ W_query, k2 = agg @ W_key ----
    gemm128(&sAgg[0][0], 128, ws + PK_Q2, &sQ2[0][0], 4, w, lane);
    gemm128(&sAgg[0][0], 128, ws + PK_K2, &sK2[0][0], 4, w, lane);
    __syncthreads();

    // ---- Phase 5: s2 = q2 @ k2^T / sqrt(128), softmax -> w2 ----
    scores32(&sQ2[0][0], &sK2[0][0], &sS2[0][0], w, lane);
    __syncthreads();
    softmax32(&sS2[0][0], &sW2h[0][0], tid);
    __syncthreads();
    for (int i = tid; i < NAG * NAG; i += 256)
        out_w2[(size_t)bid * NAG * NAG + i] = ((const float*)sS2)[i];

    // ---- Phase 6: embA = [S|act] @ W_embed, embP = [S|pol] @ W_embed (K padded 160) ----
    gemm128(&sOA[0][0], OAIP, ws + PK_EMB, &sEmbA[0][0], 5, w, lane);
    gemm128(&sOP[0][0], OAIP, ws + PK_EMB, &sEmbP[0][0], 5, w, lane);
    __syncthreads();

    // ---- Phase 7: avA = embA @ W_attn_val; delta = embP @ W_attn_val - avA ----
    gemm128(&sEmbA[0][0], 128, ws + PK_AV, &sAvA[0][0], 4, w, lane);
    __syncthreads();
    {
        const int n0 = w << 4;
        v8f c0 = {}, c1 = {};
#pragma unroll
        for (int ks = 0; ks < 4; ++ks) {
            const int k0 = ks << 5;
            v16h b  = ldB_pk(ws + PK_AV, 128, n0, k0, lane);
            v16h a0 = ldA(&sEmbP[0][0], 128, 0, k0, lane);
            v16h a1 = ldA(&sEmbP[0][0], 128, 16, k0, lane);
            c0 = wmmaf(a0, b, c0);
            c1 = wmmaf(a1, b, c1);
        }
        const int rb = (lane >> 4) << 3;
        const int cc = n0 + (lane & 15);
#pragma unroll
        for (int v = 0; v < 8; ++v)
            sDlt[rb + v][cc] = (_Float16)(c0[v] - (float)sAvA[rb + v][cc]);
#pragma unroll
        for (int v = 0; v < 8; ++v)
            sDlt[16 + rb + v][cc] = (_Float16)(c1[v] - (float)sAvA[16 + rb + v][cc]);
    }
    __syncthreads();

    // ---- Phase 8: base = w2 @ avA ----
    {
        const int n0 = w << 4;
        v16h b  = ldB_rm(&sAvA[0][0], 128, n0, 0, lane);
        v16h a0 = ldA(&sW2h[0][0], 32, 0, 0, lane);
        v16h a1 = ldA(&sW2h[0][0], 32, 16, 0, lane);
        v8f c0 = {}, c1 = {};
        c0 = wmmaf(a0, b, c0);
        c1 = wmmaf(a1, b, c1);
        stD_h(&sBase[0][0], 128, 0, n0, lane, c0);
        stD_h(&sBase[0][0], 128, 16, n0, lane, c1);
    }
    __syncthreads();

    // ---- Phase 9: P = base @ W_f1, Q = delta @ W_f1  (N=256, f32 to LDS) ----
#pragma unroll
    for (int t = 0; t < 2; ++t) {
        const int n0 = (w + (t << 3)) << 4;
        v8f p0 = {}, p1 = {}, q0 = {}, q1 = {};
        for (int ks = 0; ks < 4; ++ks) {
            const int k0 = ks << 5;
            v16h b  = ldB_pk(ws + PK_F1, 256, n0, k0, lane);
            v16h ab0 = ldA(&sBase[0][0], 128, 0, k0, lane);
            v16h ab1 = ldA(&sBase[0][0], 128, 16, k0, lane);
            v16h ad0 = ldA(&sDlt[0][0], 128, 0, k0, lane);
            v16h ad1 = ldA(&sDlt[0][0], 128, 16, k0, lane);
            p0 = wmmaf(ab0, b, p0);
            p1 = wmmaf(ab1, b, p1);
            q0 = wmmaf(ad0, b, q0);
            q1 = wmmaf(ad1, b, q1);
        }
        stD_f(&sPf[0][0], 258, 0, n0, lane, p0, 1.f);
        stD_f(&sPf[0][0], 258, 16, n0, lane, p1, 1.f);
        stD_f(&sQf[0][0], 258, 0, n0, lane, q0, 1.f);
        stD_f(&sQf[0][0], 258, 16, n0, lane, q1, 1.f);
    }
    __syncthreads();

    // ---- Phase 10: value[a,c] = sum_f leaky(P[a,f] + w2[a,c]*Q[c,f]) * Wf2[f] ----
    {
        float* ov = out_val + (size_t)bid * NAG * NAG;
#pragma unroll
        for (int r = 0; r < 4; ++r) {
            const int pi = tid + 256 * r;       // a constant per wave, c == lane
            const int a = pi >> 5, c = pi & 31;
            const float w2v = sS2[a][c];
            const float2* Pr = (const float2*)&sPf[a][0];
            const float2* Qr = (const float2*)&sQf[c][0];
            const float2* Fr = (const float2*)&sF2[0];
            float acc = 0.f;
#pragma unroll 4
            for (int i2 = 0; i2 < HID / 2; ++i2) {
                float2 p = Pr[i2], q = Qr[i2], f2 = Fr[i2];
                float t0 = fmaf(w2v, q.x, p.x);
                t0 = fmaxf(t0, 0.01f * t0);     // leaky_relu(x,0.01)
                float t1 = fmaf(w2v, q.y, p.y);
                t1 = fmaxf(t1, 0.01f * t1);
                acc = fmaf(t0, f2.x, acc);
                acc = fmaf(t1, f2.y, acc);
            }
            ov[pi] = acc;
        }
    }
#undef sAvA
#undef sDlt
#undef sEmbA
#undef sEmbP
#undef sBase
}

// Convert f32 weight [K,N] row-major -> packed u32 pairs: dst[p*N+n] = {f16(W[2p,n]), f16(W[2p+1,n])}
__global__ void pack_w_kernel(const float* __restrict__ src, unsigned int* __restrict__ dst,
                              int K, int N, int Kpad) {
    const int i = blockIdx.x * blockDim.x + threadIdx.x;
    const int total = (Kpad >> 1) * N;
    if (i >= total) return;
    const int p = i / N, n = i - p * N;
    const int k0 = 2 * p;
    const float a = (k0 < K) ? src[k0 * N + n] : 0.f;
    const float b = (k0 + 1 < K) ? src[(k0 + 1) * N + n] : 0.f;
    union { _Float16 h[2]; unsigned int u; } cv;
    cv.h[0] = (_Float16)a;
    cv.h[1] = (_Float16)b;
    dst[i] = cv.u;
}

extern "C" void kernel_launch(void* const* d_in, const int* in_sizes, int n_in,
                              void* d_out, int out_size, void* d_ws, size_t ws_size,
                              hipStream_t stream) {
    (void)in_sizes; (void)n_in; (void)out_size; (void)ws_size;
    const float* states   = (const float*)d_in[0];
    const float* policies = (const float*)d_in[1];
    const float* actions  = (const float*)d_in[2];
    const float* Wkp = (const float*)d_in[3];
    const float* Wqp = (const float*)d_in[4];
    const float* Wvp = (const float*)d_in[5];
    const float* Wk2 = (const float*)d_in[6];
    const float* Wq2 = (const float*)d_in[7];
    const float* Wem = (const float*)d_in[8];
    const float* Wav = (const float*)d_in[9];
    const float* Wf1 = (const float*)d_in[10];
    const float* Wf2 = (const float*)d_in[11];
    unsigned int* ws = (unsigned int*)d_ws;
    float* out = (float*)d_out;

    auto pk = [&](const float* s, int off, int K, int N, int Kpad) {
        const int tot = (Kpad / 2) * N;
        pack_w_kernel<<<(tot + 255) / 256, 256, 0, stream>>>(s, ws + off, K, N, Kpad);
    };
    pk(Wkp, PK_KP, 128, 128, 128);
    pk(Wqp, PK_QP, 128, 128, 128);
    pk(Wvp, PK_VP, 128, 128, 128);
    pk(Wk2, PK_K2, 128, 128, 128);
    pk(Wq2, PK_Q2, 128, 128, 128);
    pk(Wem, PK_EMB, 144, 128, 160);
    pk(Wav, PK_AV, 128, 128, 128);
    pk(Wf1, PK_F1, 128, 256, 128);

    // value [512*1024] | w1 [512*1024] | w2 [512*1024]
    critic_fused_kernel<<<512, 256, 0, stream>>>(states, policies, actions, Wf2, ws, out,
                                                 out + 512 * 1024, out + 2 * 512 * 1024);
}
